// GuidedFilter_9096740733330
// MI455X (gfx1250) — matrix operand order repaired and checked by reading, not compiled, not run on hardware
//
#include <hip/hip_runtime.h>

// Fused guided filter for MI455X (gfx1250, wave32).
// One workgroup = one 64x64 output tile per (batch, spatial tile); all box
// filtering done with V_WMMA_F32_16X16X32_F16 band-matrix multiplies using a
// hi/lo f16 split of the f32 data for near-f32 accuracy. Edge handling =
// zero-padding in LDS + analytic per-pixel window counts.

#define RAD   8
#define TILE  64
#define REG1  80            // TILE + 2R : region where A,b are needed
#define REG2  96            // TILE + 4R : region where x,y are needed
#define P2    97            // pitch for 96-wide LDS arrays
#define P1    81            // pitch for 80-wide LDS arrays
#define PH    65            // pitch for 64-wide stage-2 H-pass arrays
#define HIMG  1024
#define WIMG  1024
#define EPSF  1e-8f

typedef __attribute__((ext_vector_type(16))) _Float16 v16h;
typedef __attribute__((ext_vector_type(8)))  float    v8f;

__device__ __forceinline__ v8f wmma_f16(v16h a, v16h b, v8f c) {
  // D = A(16x32 f16) * B(32x16 f16) + C(16x16 f32)
  return __builtin_amdgcn_wmma_f32_16x16x32_f16(
      /*neg_a=*/false, a, /*neg_b=*/false, b,
      /*c_mod=*/(short)0, c, /*reuse_a=*/false, /*reuse_b=*/false);
}

struct HL { _Float16 hi, lo; };

__device__ __forceinline__ HL split_hl(float f) {
  HL r;
  r.hi = (_Float16)f;
  r.lo = (_Float16)(f - (float)r.hi);
  return r;
}

__device__ __forceinline__ float cnt1d(int i) {
  int lo = i - RAD; if (lo < 0) lo = 0;
  int hi = i + RAD; if (hi > HIMG - 1) hi = HIMG - 1;
  int c = hi - lo + 1;
  return (c < 1) ? 1.0f : (float)c;   // clamp for out-of-image (masked later)
}

__global__ __launch_bounds__(256) void guided_filter_kernel(
    const float* __restrict__ x, const float* __restrict__ y,
    float* __restrict__ out) {
  extern __shared__ float smem[];
  float* sx  = smem;                  // REG2 x P2 : x tile (zero padded)
  float* sy  = sx  + REG2 * P2;       // REG2 x P2 : y channel tile
  float* sh1 = sy  + REG2 * P2;       // REG2 x P1 : H-pass scratch 1
  float* sh2 = sh1 + REG2 * P1;       // REG2 x P1 : H-pass scratch 2
  float* mx  = sh2 + REG2 * P1;       // REG1 x P1 : mean_x
  float* vxv = mx  + REG1 * P1;       // REG1 x P1 : var_x + eps
  float* Aa  = vxv + REG1 * P1;       // REG1 x P1 : A (per channel)
  float* Bb  = Aa  + REG1 * P1;       // REG1 x P1 : b (per channel)

  const int tid  = threadIdx.x;
  const int lane = tid & 31;
  const int wave = tid >> 5;
  const int g    = lane >> 4;         // lane group (0: lanes 0-15, 1: 16-31)
  const int nn   = lane & 15;

  const int n0  = blockIdx.z;
  const int th0 = blockIdx.y * TILE;
  const int tw0 = blockIdx.x * TILE;

  // Band matrices (exact 0/1 in f16).
  // bandB: B operand for H-pass. B[k][n] = 1 iff n <= k <= n+16.
  //        B layout: lane col n = lane&15, e<8 -> k=e+8g, e>=8 -> k=8+e+8g.
  // bandA: A operand for V-pass. A[m][k] = 1 iff m <= k <= m+16.
  //        A layout: lane row m = lane&15, element e holds k = e + 16*g.
  v16h bandB, bandA;
#pragma unroll
  for (int e = 0; e < 16; ++e) {
    int kB = (e < 8) ? (e + 8 * g) : (8 + e + 8 * g);
    bandB[e] = (kB >= nn && kB <= nn + 16) ? (_Float16)1.0f : (_Float16)0.0f;
    int kA = e + 16 * g;
    bandA[e] = (kA >= nn && kA <= nn + 16) ? (_Float16)1.0f : (_Float16)0.0f;
  }

  // ---- load x tile into LDS, zero padded outside the image ----
  const float* xbase = x + (size_t)n0 * HIMG * WIMG;
  for (int idx = tid; idx < REG2 * REG2; idx += 256) {
    int r = idx / REG2, c = idx - r * REG2;
    int gh = th0 - 2 * RAD + r, gw = tw0 - 2 * RAD + c;
    float val = 0.0f;
    if (gh >= 0 && gh < HIMG && gw >= 0 && gw < WIMG)
      val = xbase[(size_t)gh * WIMG + gw];
    sx[r * P2 + c] = val;
  }
  __syncthreads();

  // ---- stage-1 H-pass for x stats: sh1 = Hbox(x), sh2 = Hbox(x*x) ----
  for (int t = wave; t < 6 * 5; t += 8) {
    int tr = t / 5, tc = t - tr * 5;
    int ro = tr * 16, co = tc * 16;
    const float* p = sx + (ro + nn) * P2 + co;   // A-operand row = lane&15
    v16h a1h, a1l, a2h, a2l;
#pragma unroll
    for (int e = 0; e < 16; ++e) {
      int K = (e < 8) ? (e + 8 * g) : (8 + e + 8 * g);
      float f  = p[K];
      float f2 = f * f;
      HL s1 = split_hl(f);  a1h[e] = s1.hi; a1l[e] = s1.lo;
      HL s2 = split_hl(f2); a2h[e] = s2.hi; a2l[e] = s2.lo;
    }
    v8f c1 = {}; c1 = wmma_f16(a1h, bandB, c1); c1 = wmma_f16(a1l, bandB, c1);
    v8f c2 = {}; c2 = wmma_f16(a2h, bandB, c2); c2 = wmma_f16(a2l, bandB, c2);
#pragma unroll
    for (int v = 0; v < 8; ++v) {
      int dm = v + 8 * g;                        // D row
      sh1[(ro + dm) * P1 + co + nn] = c1[v];
      sh2[(ro + dm) * P1 + co + nn] = c2[v];
    }
  }
  __syncthreads();

  // ---- stage-1 V-pass -> mean_x, var_x+eps on 80x80 region ----
  for (int t = wave; t < 5 * 5; t += 8) {
    int tr = t / 5, tc = t - tr * 5;
    int ro = tr * 16, co = tc * 16;
    v16h b1h, b1l, b2h, b2l;
#pragma unroll
    for (int e = 0; e < 16; ++e) {
      int k = e + 16 * g;                        // B-operand K index
      HL s1 = split_hl(sh1[(ro + k) * P1 + co + nn]);
      HL s2 = split_hl(sh2[(ro + k) * P1 + co + nn]);
      b1h[e] = s1.hi; b1l[e] = s1.lo;
      b2h[e] = s2.hi; b2l[e] = s2.lo;
    }
    v8f c1 = {}; c1 = wmma_f16(bandA, b1h, c1); c1 = wmma_f16(bandA, b1l, c1);
    v8f c2 = {}; c2 = wmma_f16(bandA, b2h, c2); c2 = wmma_f16(bandA, b2l, c2);
#pragma unroll
    for (int v = 0; v < 8; ++v) {
      int i = ro + v + 8 * g, j = co + nn;       // region-1 coords
      int gh = th0 - RAD + i, gw = tw0 - RAD + j;
      float Nrec = 1.0f / (cnt1d(gh) * cnt1d(gw));
      float mv = c1[v] * Nrec;
      mx [i * P1 + j] = mv;
      vxv[i * P1 + j] = c2[v] * Nrec - mv * mv + EPSF;
    }
  }
  __syncthreads();

  // ---- per-channel guided filter ----
  for (int ch = 0; ch < 3; ++ch) {
    const float* ybase = y + ((size_t)n0 * 3 + ch) * HIMG * WIMG;
    for (int idx = tid; idx < REG2 * REG2; idx += 256) {
      int r = idx / REG2, c = idx - r * REG2;
      int gh = th0 - 2 * RAD + r, gw = tw0 - 2 * RAD + c;
      float val = 0.0f;
      if (gh >= 0 && gh < HIMG && gw >= 0 && gw < WIMG)
        val = ybase[(size_t)gh * WIMG + gw];
      sy[r * P2 + c] = val;
    }
    __syncthreads();

    // H-pass: sh1 = Hbox(y), sh2 = Hbox(x*y)
    for (int t = wave; t < 6 * 5; t += 8) {
      int tr = t / 5, tc = t - tr * 5;
      int ro = tr * 16, co = tc * 16;
      const float* py = sy + (ro + nn) * P2 + co;
      const float* px = sx + (ro + nn) * P2 + co;
      v16h a1h, a1l, a2h, a2l;
#pragma unroll
      for (int e = 0; e < 16; ++e) {
        int K = (e < 8) ? (e + 8 * g) : (8 + e + 8 * g);
        float fy  = py[K];
        float fxy = px[K] * fy;
        HL s1 = split_hl(fy);  a1h[e] = s1.hi; a1l[e] = s1.lo;
        HL s2 = split_hl(fxy); a2h[e] = s2.hi; a2l[e] = s2.lo;
      }
      v8f c1 = {}; c1 = wmma_f16(a1h, bandB, c1); c1 = wmma_f16(a1l, bandB, c1);
      v8f c2 = {}; c2 = wmma_f16(a2h, bandB, c2); c2 = wmma_f16(a2l, bandB, c2);
#pragma unroll
      for (int v = 0; v < 8; ++v) {
        int dm = v + 8 * g;
        sh1[(ro + dm) * P1 + co + nn] = c1[v];
        sh2[(ro + dm) * P1 + co + nn] = c2[v];
      }
    }
    __syncthreads();

    // V-pass -> A, b on 80x80 region (zeroed outside the image)
    for (int t = wave; t < 5 * 5; t += 8) {
      int tr = t / 5, tc = t - tr * 5;
      int ro = tr * 16, co = tc * 16;
      v16h b1h, b1l, b2h, b2l;
#pragma unroll
      for (int e = 0; e < 16; ++e) {
        int k = e + 16 * g;
        HL s1 = split_hl(sh1[(ro + k) * P1 + co + nn]);
        HL s2 = split_hl(sh2[(ro + k) * P1 + co + nn]);
        b1h[e] = s1.hi; b1l[e] = s1.lo;
        b2h[e] = s2.hi; b2l[e] = s2.lo;
      }
      v8f cy  = {}; cy  = wmma_f16(bandA, b1h, cy);  cy  = wmma_f16(bandA, b1l, cy);
      v8f cxy = {}; cxy = wmma_f16(bandA, b2h, cxy); cxy = wmma_f16(bandA, b2l, cxy);
#pragma unroll
      for (int v = 0; v < 8; ++v) {
        int i = ro + v + 8 * g, j = co + nn;
        int gh = th0 - RAD + i, gw = tw0 - RAD + j;
        float Nrec = 1.0f / (cnt1d(gh) * cnt1d(gw));
        float my  = cy[v] * Nrec;
        float mxs = mx[i * P1 + j];
        float cov = cxy[v] * Nrec - mxs * my;
        float Av  = cov / vxv[i * P1 + j];
        float bv  = my - Av * mxs;
        bool inb = (gh >= 0) && (gh < HIMG) && (gw >= 0) && (gw < WIMG);
        Aa[i * P1 + j] = inb ? Av : 0.0f;
        Bb[i * P1 + j] = inb ? bv : 0.0f;
      }
    }
    __syncthreads();

    // stage-2 H-pass: sh1 = Hbox(A), sh2 = Hbox(b)  (80 rows x 64 cols)
    for (int t = wave; t < 5 * 4; t += 8) {
      int tr = t / 4, tc = t - tr * 4;
      int ro = tr * 16, co = tc * 16;
      const float* pa = Aa + (ro + nn) * P1 + co;
      const float* pb = Bb + (ro + nn) * P1 + co;
      v16h a1h, a1l, a2h, a2l;
#pragma unroll
      for (int e = 0; e < 16; ++e) {
        int K = (e < 8) ? (e + 8 * g) : (8 + e + 8 * g);
        HL s1 = split_hl(pa[K]); a1h[e] = s1.hi; a1l[e] = s1.lo;
        HL s2 = split_hl(pb[K]); a2h[e] = s2.hi; a2l[e] = s2.lo;
      }
      v8f c1 = {}; c1 = wmma_f16(a1h, bandB, c1); c1 = wmma_f16(a1l, bandB, c1);
      v8f c2 = {}; c2 = wmma_f16(a2h, bandB, c2); c2 = wmma_f16(a2l, bandB, c2);
#pragma unroll
      for (int v = 0; v < 8; ++v) {
        int dm = v + 8 * g;
        sh1[(ro + dm) * PH + co + nn] = c1[v];
        sh2[(ro + dm) * PH + co + nn] = c2[v];
      }
    }
    __syncthreads();

    // stage-2 V-pass -> out = mean_A * x + mean_b  (64x64 tile)
    float* obase = out + ((size_t)n0 * 3 + ch) * HIMG * WIMG;
    for (int t = wave; t < 4 * 4; t += 8) {
      int tr = t >> 2, tc = t & 3;
      int ro = tr * 16, co = tc * 16;
      v16h b1h, b1l, b2h, b2l;
#pragma unroll
      for (int e = 0; e < 16; ++e) {
        int k = e + 16 * g;
        HL s1 = split_hl(sh1[(ro + k) * PH + co + nn]);
        HL s2 = split_hl(sh2[(ro + k) * PH + co + nn]);
        b1h[e] = s1.hi; b1l[e] = s1.lo;
        b2h[e] = s2.hi; b2l[e] = s2.lo;
      }
      v8f cA = {}; cA = wmma_f16(bandA, b1h, cA); cA = wmma_f16(bandA, b1l, cA);
      v8f cB = {}; cB = wmma_f16(bandA, b2h, cB); cB = wmma_f16(bandA, b2l, cB);
#pragma unroll
      for (int v = 0; v < 8; ++v) {
        int i = ro + v + 8 * g, j = co + nn;     // 0..63 in tile
        int gh = th0 + i, gw = tw0 + j;          // always in-bounds (1024%64==0)
        float Nrec = 1.0f / (cnt1d(gh) * cnt1d(gw));
        float xvv  = sx[(i + 2 * RAD) * P2 + (j + 2 * RAD)];
        obase[(size_t)gh * WIMG + gw] = cA[v] * Nrec * xvv + cB[v] * Nrec;
      }
    }
    __syncthreads();
  }
}

extern "C" void kernel_launch(void* const* d_in, const int* in_sizes, int n_in,
                              void* d_out, int out_size, void* d_ws, size_t ws_size,
                              hipStream_t stream) {
  (void)n_in; (void)out_size; (void)d_ws; (void)ws_size;
  const float* x = (const float*)d_in[0];
  const float* y = (const float*)d_in[1];
  float* out = (float*)d_out;
  int nbatch = in_sizes[0] / (HIMG * WIMG);      // x is [N,1,H,W]
  dim3 grid(WIMG / TILE, HIMG / TILE, nbatch);
  size_t shmem =
      (size_t)(2 * REG2 * P2 + 2 * REG2 * P1 + 4 * REG1 * P1) * sizeof(float);
  hipLaunchKernelGGL(guided_filter_kernel, grid, dim3(256), shmem, stream,
                     x, y, out);
}
